// OrkySparseGlobalHead_19258633356049
// MI455X (gfx1250) — matrix operand order repaired and checked by reading, not compiled, not run on hardware
//
#include <hip/hip_runtime.h>

// ---------------------------------------------------------------------------
// Sparse global attention (single head) for MI455X / gfx1250, wave32 + WMMA
// + Tensor Data Mover staging of the K panel into LDS.
// ---------------------------------------------------------------------------

typedef __attribute__((ext_vector_type(16))) _Float16 v16h;
typedef __attribute__((ext_vector_type(8)))  float    v8f;

union AV { v16h v; _Float16 e[16]; uint4 q[2]; };
union CV { v8f  v; float     e[8];  };

constexpr int   Bn    = 4;
constexpr int   S     = 4096;
constexpr int   Dm    = 1024;
constexpr int   Hd    = 64;
constexpr int   BS    = Bn * S;       // 16384 total rows
constexpr int   KSP   = 409;          // max(1, int(4096*0.1))
constexpr float SCALE = 0.125f;       // 64^-0.5

// LDS layout for attn_kernel (dynamic shared memory)
constexpr int SC_OFF     = 0;                        // 16*4096 f32 score slab
constexpr int HIST_OFF   = 16 * S * 4;               // 8 waves * 256 bins u32
constexpr int WSEL_OFF   = HIST_OFF + 8 * 256 * 4;   // 16 u32 per-wave prefix/rem
constexpr int RKEY_OFF   = WSEL_OFF + 16 * 4;        // 16 u32 row threshold keys
constexpr int OACC_OFF   = RKEY_OFF + 16 * 4;        // 16*64 f32 output acc
constexpr int STAGE_OFF  = OACC_OFF + 16 * Hd * 4;   // K staging (double buffer)
constexpr int STAGE_HALF = 128 * Hd * 2;             // 16KB: 128 keys x 64 f16
constexpr int SMEM_BYTES = STAGE_OFF + 2 * STAGE_HALF;  // 307,328 B <= 320KB

__device__ inline unsigned fkey(float f) {
  unsigned b = __float_as_uint(f);
  return b ^ ((b & 0x80000000u) ? 0xFFFFFFFFu : 0x80000000u); // order-preserving
}

__device__ inline void cvt8(AV& a, int base, float4 u, float4 w) {
  a.e[base + 0] = (_Float16)u.x; a.e[base + 1] = (_Float16)u.y;
  a.e[base + 2] = (_Float16)u.z; a.e[base + 3] = (_Float16)u.w;
  a.e[base + 4] = (_Float16)w.x; a.e[base + 5] = (_Float16)w.y;
  a.e[base + 6] = (_Float16)w.z; a.e[base + 7] = (_Float16)w.w;
}

// ---------------------------------------------------------------------------
// Tensor Data Mover: stage 128 rows x 64 f16 of the K panel into LDS.
// D# built per CDNA5 ISA ch.8: 2D tile, data_size=2B, dim0=64, stride0=64.
// This toolchain exposes the 6-arg builtin (extra i32x8 group before cpol).
// ---------------------------------------------------------------------------
#if __has_builtin(__builtin_amdgcn_tensor_load_to_lds)
#define HAVE_TDM 1
typedef __attribute__((ext_vector_type(4))) unsigned u32x4;
typedef __attribute__((ext_vector_type(8))) int      i32x8;
typedef __attribute__((ext_vector_type(4))) int      i32x4;

__device__ inline void tdm_load_k_chunk(const _Float16* gsrc, unsigned lds_off) {
  const unsigned long long ga = (unsigned long long)(uintptr_t)gsrc;
  u32x4 g0;
  g0[0] = 1u;                                   // count=1, user descriptor
  g0[1] = lds_off;                              // lds_addr (bytes)
  g0[2] = (unsigned)(ga & 0xFFFFFFFFu);         // global_addr[31:0]
  g0[3] = (unsigned)((ga >> 32) & 0x1FFFFFFu)   // global_addr[56:32]
        | (2u << 30);                           // type=2 ("image")
  i32x8 g1;
  g1[0] = (int)(1u << 16);                      // wg_mask=0, data_size=1 (2B)
  g1[1] = (int)(64u << 16);                     // tensor_dim0 = 64
  g1[2] = (int)(4096u << 16);                   // tensor_dim1 = 4096
  g1[3] = (int)(64u << 16);                     // tile_dim0 = 64
  g1[4] = (int)128;                             // tile_dim1 = 128 rows
  g1[5] = (int)64;                              // tensor_dim0_stride = 64
  g1[6] = 0;
  g1[7] = 0;
  i32x4 gz4 = {0, 0, 0, 0};                     // groups 2/3 unused (2D tensor)
  i32x8 gz8 = {0, 0, 0, 0, 0, 0, 0, 0};         // trailing group (clang-23 form)
  __builtin_amdgcn_tensor_load_to_lds(g0, g1, gz4, gz4, gz8, 0);
}
#else
#define HAVE_TDM 0
#endif

// ---------------------------------------------------------------------------
// Kernel 1: qkv = x @ W^T -> q [BS,64] f16, k [BS,64] f16, vT [64,BS] f16.
// One wave per 16x16 tile; 32 x v_wmma_f32_16x16x32_f16 over K=1024.
// Destination (q/k/vT) is uniform per block (by ntile) -> branch-free stores.
// ---------------------------------------------------------------------------
__global__ __launch_bounds__(256) void qkv_kernel(const float* __restrict__ x,
                                                  const float* __restrict__ w,
                                                  _Float16* __restrict__ q,
                                                  _Float16* __restrict__ k,
                                                  _Float16* __restrict__ vT) {
  const int wave  = threadIdx.x >> 5;
  const int lane  = threadIdx.x & 31;
  const int half  = lane >> 4;
  const int lr    = lane & 15;
  const int mtile = blockIdx.x * 8 + wave;       // 1024 m-tiles
  const int ntile = blockIdx.y;                  // 12 n-tiles (192 cols)
  const int row   = mtile * 16 + lr;             // A row (x row)
  const int ncol  = ntile * 16 + lr;             // B column (W row)

  const float* xrow = x + (size_t)row * Dm;
  const float* wrow = w + (size_t)ncol * Dm;

  CV c = {};
  for (int ks = 0; ks < Dm / 32; ++ks) {
    AV a, b;
    // A 16x32 f16: lane holds two contiguous 8-elem runs (layout-derived)
    const int abase = ks * 32 + half * 8;
    cvt8(a, 0, *(const float4*)(xrow + abase),     *(const float4*)(xrow + abase + 4));
    cvt8(a, 8, *(const float4*)(xrow + abase + 16),*(const float4*)(xrow + abase + 20));
    // B 32x16 f16: lane holds two contiguous 8-elem runs (same k order as A)
    const int bbase = ks * 32 + half * 8;
    cvt8(b, 0, *(const float4*)(wrow + bbase),     *(const float4*)(wrow + bbase + 4));
    cvt8(b, 8, *(const float4*)(wrow + bbase + 16), *(const float4*)(wrow + bbase + 20));
    c.v = __builtin_amdgcn_wmma_f32_16x16x32_f16(false, a.v, false, b.v,
                                                 (short)0, c.v, false, false);
  }

  // Uniform destination select (ntile is block-uniform): no divergent stores.
  if (ntile < 4) {
#pragma unroll
    for (int i = 0; i < 8; ++i)
      q[(size_t)(mtile * 16 + i + 8 * half) * Hd + ncol] = (_Float16)c.e[i];
  } else if (ntile < 8) {
#pragma unroll
    for (int i = 0; i < 8; ++i)
      k[(size_t)(mtile * 16 + i + 8 * half) * Hd + (ncol - 64)] = (_Float16)c.e[i];
  } else {
#pragma unroll
    for (int i = 0; i < 8; ++i)   // transposed store: vT[col][row]
      vT[(size_t)(ncol - 128) * BS + (mtile * 16 + i + 8 * half)] = (_Float16)c.e[i];
  }
}

// ---------------------------------------------------------------------------
// Kernel 2: per (batch, 16-query tile): QK^T scores -> 256KB LDS slab
// (K panel TDM-streamed through a 2x16KB LDS double buffer), per-row radix
// select of the top-409 threshold, softmax in LDS, then PV via WMMA.
// ---------------------------------------------------------------------------
__global__ __launch_bounds__(256) void attn_kernel(const _Float16* __restrict__ q,
                                                   const _Float16* __restrict__ k,
                                                   const _Float16* __restrict__ vT,
                                                   float* __restrict__ out) {
  extern __shared__ __align__(16) char smem[];
  float*    sc     = (float*)(smem + SC_OFF);
  unsigned* hist   = (unsigned*)(smem + HIST_OFF);
  unsigned* wsel   = (unsigned*)(smem + WSEL_OFF);
  unsigned* rowkey = (unsigned*)(smem + RKEY_OFF);
  float*    oacc   = (float*)(smem + OACC_OFF);

  const int wave = threadIdx.x >> 5;
  const int lane = threadIdx.x & 31;
  const int half = lane >> 4;
  const int lr   = lane & 15;
  const int qt   = blockIdx.x;                   // query tile
  const int b    = blockIdx.y;                   // batch

  const _Float16* qb = q + (size_t)b * S * Hd;
  const _Float16* kb = k + (size_t)b * S * Hd;

  for (int i = threadIdx.x; i < 16 * Hd; i += 256) oacc[i] = 0.0f;

  // ---- Phase 1: scores = scale * q @ k^T (K streamed via TDM double buffer)
  AV a0, a1;
  {
    const _Float16* qrow = qb + (size_t)(qt * 16 + lr) * Hd;
    a0.q[0] = *(const uint4*)(qrow + half * 8);
    a0.q[1] = *(const uint4*)(qrow + 16 + half * 8);
    a1.q[0] = *(const uint4*)(qrow + 32 + half * 8);
    a1.q[1] = *(const uint4*)(qrow + 48 + half * 8);
  }
#if HAVE_TDM
  if (wave == 0) tdm_load_k_chunk(kb, STAGE_OFF);     // prime chunk 0
#endif
  for (int ch = 0; ch < S / 128; ++ch) {              // 32 chunks of 128 keys
    char* stg = smem + STAGE_OFF + (ch & 1) * STAGE_HALF;
#if HAVE_TDM
    if (wave == 0) {
      if (ch + 1 < S / 128) {                         // overlap next DMA
        tdm_load_k_chunk(kb + (size_t)(ch + 1) * 128 * Hd,
                         (unsigned)(STAGE_OFF + ((ch + 1) & 1) * STAGE_HALF));
        __builtin_amdgcn_s_wait_tensorcnt(1);         // chunk ch landed
      } else {
        __builtin_amdgcn_s_wait_tensorcnt(0);         // last chunk landed
      }
    }
#else
    {
      const _Float16* srcc = kb + (size_t)ch * 128 * Hd;
      for (int i = threadIdx.x; i < STAGE_HALF / 16; i += 256)
        ((uint4*)stg)[i] = ((const uint4*)srcc)[i];
    }
#endif
    __syncthreads();                                  // staged data visible

    // Each wave computes one 16-key tile of this chunk from LDS.
    const int t = ch * 8 + wave;                      // global key tile
    const char* krow = stg + (wave * 16 + lr) * (Hd * 2);
    AV b0, b1;
    b0.q[0] = *(const uint4*)(krow + half * 16);
    b0.q[1] = *(const uint4*)(krow + 32 + half * 16);
    b1.q[0] = *(const uint4*)(krow + 64 + half * 16);
    b1.q[1] = *(const uint4*)(krow + 64 + 32 + half * 16);
    CV c = {};
    c.v = __builtin_amdgcn_wmma_f32_16x16x32_f16(false, a0.v, false, b0.v,
                                                 (short)0, c.v, false, false);
    c.v = __builtin_amdgcn_wmma_f32_16x16x32_f16(false, a1.v, false, b1.v,
                                                 (short)0, c.v, false, false);
#pragma unroll
    for (int i = 0; i < 8; ++i)
      sc[(i + 8 * half) * S + t * 16 + lr] = c.e[i] * SCALE;

    __syncthreads();                                  // compute done before
  }                                                   // buffer is re-filled

  // ---- Phase 2: per-row descending radix select of the 409th-largest key
  unsigned* h = hist + wave * 256;
  for (int rr = 0; rr < 2; ++rr) {                    // each wave owns 2 rows
    const int row = wave * 2 + rr;
    unsigned prefix = 0, rem = KSP;
    for (int p = 0; p < 4; ++p) {
      for (int i = lane; i < 256; i += 32) h[i] = 0;
      __syncthreads();
      const unsigned maskHi = p ? ~(0xFFFFFFFFu >> (8 * p)) : 0u;
      const int shift = 24 - 8 * p;
      for (int i = lane; i < S; i += 32) {
        const unsigned key = fkey(sc[row * S + i]);
        if ((key & maskHi) == prefix)
          atomicAdd(&h[(key >> shift) & 0xFF], 1u);
      }
      __syncthreads();
      if (lane == 0) {
        unsigned cum = 0; int d = 0;
        for (int bin = 255; bin >= 0; --bin) {
          cum += h[bin];
          if (cum >= rem) { d = bin; rem -= (cum - h[bin]); break; }
        }
        prefix |= (unsigned)d << shift;
        wsel[wave * 2]     = prefix;
        wsel[wave * 2 + 1] = rem;
      }
      __syncthreads();
      prefix = wsel[wave * 2];
      rem    = wsel[wave * 2 + 1];
    }
    if (lane == 0) rowkey[row] = prefix;
  }
  __syncthreads();

  // ---- Phase 3: masked softmax in-place (kept iff key >= threshold key)
  for (int rr = 0; rr < 2; ++rr) {
    const int row = wave * 2 + rr;
    const unsigned keyT = rowkey[row];
    float mx = -3.0e38f;
    for (int i = lane; i < S; i += 32) {
      const float s = sc[row * S + i];
      if (fkey(s) >= keyT) mx = fmaxf(mx, s);
    }
#pragma unroll
    for (int off = 16; off > 0; off >>= 1) mx = fmaxf(mx, __shfl_xor(mx, off, 32));
    float sum = 0.0f;
    for (int i = lane; i < S; i += 32) {
      const float s = sc[row * S + i];
      if (fkey(s) >= keyT) sum += __expf(s - mx);
    }
#pragma unroll
    for (int off = 16; off > 0; off >>= 1) sum += __shfl_xor(sum, off, 32);
    const float inv = 1.0f / sum;
    for (int i = lane; i < S; i += 32) {
      const float s = sc[row * S + i];
      sc[row * S + i] = (fkey(s) >= keyT) ? __expf(s - mx) * inv : 0.0f;
    }
  }
  __syncthreads();

  // ---- Phase 4: out = probs @ v  (8 waves = 4 n-tiles x 2 K-halves)
  {
    const int ntile = wave & 3;
    const int khalf = wave >> 2;
    const int vcol  = ntile * 16 + lr;
    const _Float16* vtb = vT + (size_t)vcol * BS + (size_t)b * S;  // v^T row
    const float* scrow = sc + lr * S;
    CV c = {};
    for (int ks = 0; ks < 64; ++ks) {
      const int kbase = khalf * 2048 + ks * 32;
      AV a, bm;
      // A: probs f32 -> f16, two contiguous 8-float runs per lane (LDS b128)
      cvt8(a, 0, *(const float4*)(scrow + kbase + half * 8),
                 *(const float4*)(scrow + kbase + half * 8 + 4));
      cvt8(a, 8, *(const float4*)(scrow + kbase + 16 + half * 8),
                 *(const float4*)(scrow + kbase + 16 + half * 8 + 4));
      // B: v^T makes K-values contiguous -> two 8-elem b128 loads (same k order as A)
      bm.q[0] = *(const uint4*)(vtb + kbase + half * 8);
      bm.q[1] = *(const uint4*)(vtb + kbase + 16 + half * 8);
      if (kbase + 64 < S)
        __builtin_prefetch(vtb + kbase + 64, 0, 0);
      c.v = __builtin_amdgcn_wmma_f32_16x16x32_f16(false, a.v, false, bm.v,
                                                   (short)0, c.v, false, false);
    }
#pragma unroll
    for (int i = 0; i < 8; ++i)
      atomicAdd(&oacc[(i + 8 * half) * Hd + vcol], c.e[i]);  // join 2 K-halves
  }
  __syncthreads();

  for (int i = threadIdx.x; i < 16 * Hd; i += 256) {
    const int m = i >> 6, n = i & 63;
    out[((size_t)b * S + qt * 16 + m) * Hd + n] = oacc[i];
  }
}

// ---------------------------------------------------------------------------
extern "C" void kernel_launch(void* const* d_in, const int* in_sizes, int n_in,
                              void* d_out, int out_size, void* d_ws, size_t ws_size,
                              hipStream_t stream) {
  const float* x = (const float*)d_in[0];     // [4,4096,1024] f32
  const float* w = (const float*)d_in[1];     // [192,1024] f32
  float* out = (float*)d_out;                 // [4,4096,64] f32

  _Float16* qb  = (_Float16*)d_ws;            // 2 MB
  _Float16* kb  = qb + (size_t)BS * Hd;       // 2 MB
  _Float16* vTb = kb + (size_t)BS * Hd;       // 2 MB (transposed [64][BS])

  (void)in_sizes; (void)n_in; (void)out_size; (void)ws_size;

  (void)hipFuncSetAttribute(reinterpret_cast<const void*>(attn_kernel),
                            hipFuncAttributeMaxDynamicSharedMemorySize,
                            SMEM_BYTES);

  // QKV projection: 1024 m-tiles / 8 waves per block, 12 n-tiles
  qkv_kernel<<<dim3(BS / 16 / 8, (3 * Hd) / 16), 256, 0, stream>>>(
      x, w, qb, kb, vTb);

  // Attention: one block per (query tile, batch), 300KB LDS working set
  attn_kernel<<<dim3(S / 16, Bn), 256, SMEM_BYTES, stream>>>(qb, kb, vTb, out);
}
